// Attention_13099650253523
// MI455X (gfx1250) — matrix operand order repaired and checked by reading, not compiled, not run on hardware
//
#include <hip/hip_runtime.h>
#include <math.h>

#define B_  16
#define N_  1024
#define C_  768
#define H_  12
#define HD_ 64
static constexpr int   DQKV       = 3 * C_;   // 2304
static constexpr float ATTN_SCALE = 0.125f;   // 64^-0.5

typedef _Float16 v16h __attribute__((ext_vector_type(16)));
typedef _Float16 h8   __attribute__((ext_vector_type(8)));
typedef float    v8f  __attribute__((ext_vector_type(8)));

// ---------------------------------------------------------------------------
// WMMA fragment loaders (layouts per CDNA5 ISA, wave32)
// A 16x32 f16: lane<16 -> row M=lane, K = {0..7, 16..23}; lane>=16 -> row
// M=lane-16, K = {8..15, 24..31}.  (two contiguous 16B chunks)
// ---------------------------------------------------------------------------
__device__ __forceinline__ v16h wmma_a_load(const _Float16* base, int ld,
                                            int row0, int k0, int lane) {
  const int m = row0 + (lane & 15);
  const _Float16* p = base + (size_t)m * ld + k0 + ((lane & 16) ? 8 : 0);
  h8 lo = *(const h8*)(p);
  h8 hi = *(const h8*)(p + 16);
  v16h a;
#pragma unroll
  for (int i = 0; i < 8; ++i) { a[i] = lo[i]; a[i + 8] = hi[i]; }
  return a;
}

// B 32x16 f16: lane<16 -> col N=lane, K = {0..7, 16..23}; lane>=16 -> col
// N=lane-16, K = {8..15, 24..31}.  (two contiguous 16B chunks)
__device__ __forceinline__ v16h wmma_b_load(const _Float16* base, int ld,
                                            int col0, int k0, int lane) {
  const int n = col0 + (lane & 15);
  const _Float16* p = base + (size_t)n * ld + k0 + ((lane & 16) ? 8 : 0);
  h8 lo = *(const h8*)(p);
  h8 hi = *(const h8*)(p + 16);
  v16h b;
#pragma unroll
  for (int i = 0; i < 8; ++i) { b[i] = lo[i]; b[i + 8] = hi[i]; }
  return b;
}

// ---------------------------------------------------------------------------
// Scale / quantize / convert helpers
// ---------------------------------------------------------------------------
__global__ void zero2_kernel(float* p) {
  if (threadIdx.x < 2) p[threadIdx.x] = 0.0f;
}

__global__ void abssum_kernel(const float* __restrict__ w, long n,
                              float* __restrict__ out) {
  __shared__ float red[256];
  float acc = 0.0f;
  for (long i = (long)blockIdx.x * 256 + threadIdx.x; i < n;
       i += (long)gridDim.x * 256)
    acc += fabsf(w[i]);
  red[threadIdx.x] = acc;
  __syncthreads();
  for (int s = 128; s > 0; s >>= 1) {
    if (threadIdx.x < s) red[threadIdx.x] += red[threadIdx.x + s];
    __syncthreads();
  }
  if (threadIdx.x == 0) atomicAdd(out, red[0]);
}

// ternary codes {-1,0,+1} as f16; scale applied later in GEMM epilogues
__global__ void quant_kernel(const float* __restrict__ w,
                             _Float16* __restrict__ t, long n,
                             const float* __restrict__ sum_ptr,
                             float inv_numel) {
  const float s   = sum_ptr[0] * inv_numel;      // mean(|W|)
  const float inv = 1.0f / (s + 1e-5f);
  for (long i = (long)blockIdx.x * 256 + threadIdx.x; i < n;
       i += (long)gridDim.x * 256)
    t[i] = (_Float16)rintf(fminf(fmaxf(w[i] * inv, -1.0f), 1.0f));
}

__global__ void cvt_kernel(const float* __restrict__ x,
                           _Float16* __restrict__ y, long n) {
  for (long i = (long)blockIdx.x * 256 + threadIdx.x; i < n;
       i += (long)gridDim.x * 256)
    y[i] = (_Float16)x[i];
}

// ---------------------------------------------------------------------------
// QKV GEMM: qkv[t, d] = s_qkv * sum_c x16[t, c] * T[d, c]
// Each wave computes a 16x64 strip (4 N-tiles share one A fragment):
// 4 wmma per A load -> WMMA-bound instead of VMEM-issue-bound.
// Q,K stored [B,H,N,64]; V stored transposed [B,H,64,N].
// ---------------------------------------------------------------------------
__global__ void __launch_bounds__(256)
qkv_gemm_kernel(const _Float16* __restrict__ x16,
                const _Float16* __restrict__ tw,
                _Float16* __restrict__ qb, _Float16* __restrict__ kbf,
                _Float16* __restrict__ vt,
                const float* __restrict__ sum_ptr, float inv_numel) {
  const int lane = threadIdx.x & 31;
  const int wave = threadIdx.x >> 5;
  const int tile = blockIdx.x * 8 + wave;
  const int nt = tile % (DQKV / 64);    // 36 strips of 64 cols
  const int mt = tile / (DQKV / 64);
  const int row0 = mt * 16, col0 = nt * 64;
  const float s = sum_ptr[0] * inv_numel;

  v8f acc[4] = {};
#pragma unroll 2
  for (int k0 = 0; k0 < C_; k0 += 32) {
    v16h a = wmma_a_load(x16, C_, row0, k0, lane);
    __builtin_prefetch(x16 + (size_t)(row0 + (lane & 15)) * C_ + k0 + 64, 0, 1);
#pragma unroll
    for (int c = 0; c < 4; ++c) {
      v16h b = wmma_b_load(tw, C_, col0 + c * 16, k0, lane);
      acc[c] = __builtin_amdgcn_wmma_f32_16x16x32_f16(false, a, false, b,
                                                      (short)0, acc[c],
                                                      false, false);
    }
  }

  // epilogue: 64-wide strip is uniform in `which` and head h (768%64==0)
  const int d0     = col0 + (lane & 15);
  const int which  = d0 / C_;
  const int h      = (d0 % C_) / HD_;
  const int rowoff = (lane & 16) ? 8 : 0;
#pragma unroll
  for (int c = 0; c < 4; ++c) {
    const int hd = (col0 % HD_) + c * 16 + (lane & 15);  // col0%64==0 -> c*16+lane
#pragma unroll
    for (int j = 0; j < 8; ++j) {
      const int t  = row0 + j + rowoff;
      const int b_ = t >> 10;            // / N_
      const int n  = t & (N_ - 1);
      const size_t bh = (size_t)b_ * H_ + h;
      const _Float16 val = (_Float16)(acc[c][j] * s);
      if (which == 0)      qb [(bh * N_ + n) * HD_ + hd]  = val;
      else if (which == 1) kbf[(bh * N_ + n) * HD_ + hd]  = val;
      else                 vt [(bh * HD_ + hd) * N_ + n]  = val;   // V transposed
    }
  }
}

// ---------------------------------------------------------------------------
// Flash attention: one 16-query tile per wave, stream over keys in 32-chunks.
// ---------------------------------------------------------------------------
__global__ void __launch_bounds__(256)
flash_attn_kernel(const _Float16* __restrict__ qb,
                  const _Float16* __restrict__ kbf,
                  const _Float16* __restrict__ vt,
                  _Float16* __restrict__ ao) {
  __shared__ __align__(32) _Float16 ldsP[8][16 * 32];  // per-wave P staging
  const int lane = threadIdx.x & 31;
  const int wave = threadIdx.x >> 5;
  const int bh = blockIdx.x >> 3;
  const int qc = blockIdx.x & 7;
  const int b_ = bh / H_, h = bh % H_;
  const int n0 = qc * 128 + wave * 16;     // first query token of this wave

  const _Float16* qbase = qb  + (size_t)bh * N_ * HD_;
  const _Float16* kbase = kbf + (size_t)bh * N_ * HD_;
  const _Float16* vbase = vt  + (size_t)bh * HD_ * N_;

  // Q fragments for the two 32-wide K-slices of HD=64 (loaded once)
  const v16h qa0 = wmma_a_load(qbase, HD_, n0, 0, lane);
  const v16h qa1 = wmma_a_load(qbase, HD_, n0, 32, lane);

  const int rowoff = (lane & 16) ? 8 : 0;
  float mrow[8], lrow[8];
  v8f oacc[4] = {};
#pragma unroll
  for (int j = 0; j < 8; ++j) { mrow[j] = -__builtin_huge_valf(); lrow[j] = 0.0f; }

  _Float16* myP = &ldsP[wave][0];

  for (int j0 = 0; j0 < N_; j0 += 32) {
    // ---- S = Q @ K^T for keys [j0, j0+32), two 16x16 tiles --------------
    v8f s0 = {}, s1 = {};
    {
      v16h b00 = wmma_b_load(kbase, HD_, j0, 0, lane);
      v16h b01 = wmma_b_load(kbase, HD_, j0, 32, lane);
      s0 = __builtin_amdgcn_wmma_f32_16x16x32_f16(false, qa0, false, b00,
                                                  (short)0, s0, false, false);
      s0 = __builtin_amdgcn_wmma_f32_16x16x32_f16(false, qa1, false, b01,
                                                  (short)0, s0, false, false);
      v16h b10 = wmma_b_load(kbase, HD_, j0 + 16, 0, lane);
      v16h b11 = wmma_b_load(kbase, HD_, j0 + 16, 32, lane);
      s1 = __builtin_amdgcn_wmma_f32_16x16x32_f16(false, qa0, false, b10,
                                                  (short)0, s1, false, false);
      s1 = __builtin_amdgcn_wmma_f32_16x16x32_f16(false, qa1, false, b11,
                                                  (short)0, s1, false, false);
    }

    // ---- streaming softmax (per-row stats via 16-lane xor reductions) ---
#pragma unroll
    for (int j = 0; j < 8; ++j) {
      s0[j] *= ATTN_SCALE;
      s1[j] *= ATTN_SCALE;
      float rm = fmaxf(s0[j], s1[j]);
#pragma unroll
      for (int msk = 1; msk < 16; msk <<= 1)
        rm = fmaxf(rm, __shfl_xor(rm, msk, 32));
      const float mn    = fmaxf(mrow[j], rm);
      const float alpha = __expf(mrow[j] - mn);
      const float p0 = __expf(s0[j] - mn);
      const float p1 = __expf(s1[j] - mn);
      s0[j] = p0; s1[j] = p1;
      float rs = p0 + p1;
#pragma unroll
      for (int msk = 1; msk < 16; msk <<= 1)
        rs += __shfl_xor(rs, msk, 32);
      lrow[j] = lrow[j] * alpha + rs;
      mrow[j] = mn;
#pragma unroll
      for (int c = 0; c < 4; ++c) oacc[c][j] *= alpha;
    }

    // ---- re-layout P: C-format f32 -> LDS row-major f16 [16][32] --------
#pragma unroll
    for (int j = 0; j < 8; ++j) {
      const int m = j + rowoff;
      const int n = lane & 15;
      myP[m * 32 + n]      = (_Float16)s0[j];
      myP[m * 32 + 16 + n] = (_Float16)s1[j];
    }
    asm volatile("s_wait_dscnt 0x0" ::: "memory");  // intra-wave LDS ordering

    // ---- load P back as 16x32 A fragment --------------------------------
    v16h pa;
    {
      const int m = lane & 15;
      const _Float16* p = myP + m * 32 + ((lane & 16) ? 8 : 0);
      h8 lo = *(const h8*)(p);
      h8 hi = *(const h8*)(p + 16);
#pragma unroll
      for (int i = 0; i < 8; ++i) { pa[i] = lo[i]; pa[i + 8] = hi[i]; }
    }

    // ---- O += P @ V  (V^T rows contiguous in key -> clean B loads) ------
#pragma unroll
    for (int c = 0; c < 4; ++c) {
      v16h vb = wmma_b_load(vbase, N_, c * 16, j0, lane);
      oacc[c] = __builtin_amdgcn_wmma_f32_16x16x32_f16(false, pa, false, vb,
                                                       (short)0, oacc[c],
                                                       false, false);
    }
  }

  // ---- normalize and store attention output [B,N,C] ----------------------
#pragma unroll
  for (int j = 0; j < 8; ++j) {
    const float invl = 1.0f / lrow[j];
    const int t = n0 + j + rowoff;
#pragma unroll
    for (int c = 0; c < 4; ++c) {
      const int dcol = h * HD_ + c * 16 + (lane & 15);
      ao[((size_t)b_ * N_ + t) * C_ + dcol] = (_Float16)(oacc[c][j] * invl);
    }
  }
}

// ---------------------------------------------------------------------------
// Output projection: out[t, d] = s_proj * sum_c ao[t, c] * Tp[d, c] + bias[d]
// 16x64 strip per wave, same blocking as QKV GEMM.
// ---------------------------------------------------------------------------
__global__ void __launch_bounds__(256)
proj_gemm_kernel(const _Float16* __restrict__ ao,
                 const _Float16* __restrict__ tw,
                 const float* __restrict__ bias, float* __restrict__ out,
                 const float* __restrict__ sum_ptr, float inv_numel) {
  const int lane = threadIdx.x & 31;
  const int wave = threadIdx.x >> 5;
  const int tile = blockIdx.x * 8 + wave;
  const int nt = tile % (C_ / 64);      // 12 strips of 64 cols
  const int mt = tile / (C_ / 64);
  const int row0 = mt * 16, col0 = nt * 64;
  const float s = sum_ptr[0] * inv_numel;

  v8f acc[4] = {};
#pragma unroll 2
  for (int k0 = 0; k0 < C_; k0 += 32) {
    v16h a = wmma_a_load(ao, C_, row0, k0, lane);
    __builtin_prefetch(ao + (size_t)(row0 + (lane & 15)) * C_ + k0 + 64, 0, 1);
#pragma unroll
    for (int c = 0; c < 4; ++c) {
      v16h b = wmma_b_load(tw, C_, col0 + c * 16, k0, lane);
      acc[c] = __builtin_amdgcn_wmma_f32_16x16x32_f16(false, a, false, b,
                                                      (short)0, acc[c],
                                                      false, false);
    }
  }

  const int rowoff = (lane & 16) ? 8 : 0;
#pragma unroll
  for (int c = 0; c < 4; ++c) {
    const int d  = col0 + c * 16 + (lane & 15);
    const float bd = bias[d];
#pragma unroll
    for (int j = 0; j < 8; ++j) {
      const int t = row0 + j + rowoff;
      out[(size_t)t * C_ + d] = acc[c][j] * s + bd;
    }
  }
}

// ---------------------------------------------------------------------------
extern "C" void kernel_launch(void* const* d_in, const int* in_sizes, int n_in,
                              void* d_out, int out_size, void* d_ws,
                              size_t ws_size, hipStream_t stream) {
  const float* x      = (const float*)d_in[0];
  const float* w_qkv  = (const float*)d_in[1];
  const float* w_proj = (const float*)d_in[2];
  const float* b_proj = (const float*)d_in[3];
  float* out = (float*)d_out;

  char* ws = (char*)d_ws;
  float* scales = (float*)ws;                       // [0]=sum|Wqkv| [1]=sum|Wproj|
  size_t off = 256;
  _Float16* tqkv  = (_Float16*)(ws + off); off += (size_t)DQKV * C_ * 2;
  _Float16* tproj = (_Float16*)(ws + off); off += (size_t)C_ * C_ * 2;
  _Float16* x16   = (_Float16*)(ws + off); off += (size_t)B_ * N_ * C_ * 2;
  _Float16* qb    = (_Float16*)(ws + off); off += (size_t)B_ * H_ * N_ * HD_ * 2;
  _Float16* kbf   = (_Float16*)(ws + off); off += (size_t)B_ * H_ * N_ * HD_ * 2;
  _Float16* vt    = (_Float16*)(ws + off); off += (size_t)B_ * H_ * HD_ * N_ * 2;
  _Float16* ao    = (_Float16*)(ws + off); off += (size_t)B_ * N_ * C_ * 2;

  const long n_qkv  = (long)DQKV * C_;   // 1,769,472
  const long n_proj = (long)C_ * C_;     //   589,824
  const long n_x    = (long)B_ * N_ * C_;

  zero2_kernel<<<1, 32, 0, stream>>>(scales);
  abssum_kernel<<<512, 256, 0, stream>>>(w_qkv, n_qkv, scales + 0);
  abssum_kernel<<<256, 256, 0, stream>>>(w_proj, n_proj, scales + 1);
  quant_kernel<<<512, 256, 0, stream>>>(w_qkv, tqkv, n_qkv, scales + 0,
                                        1.0f / (float)n_qkv);
  quant_kernel<<<256, 256, 0, stream>>>(w_proj, tproj, n_proj, scales + 1,
                                        1.0f / (float)n_proj);
  cvt_kernel<<<1024, 256, 0, stream>>>(x, x16, n_x);

  // (B*N/16) M-tiles x (2304/64) N-strips, 8 waves/block
  qkv_gemm_kernel<<<(1024 * 36) / 8, 256, 0, stream>>>(
      x16, tqkv, qb, kbf, vt, scales + 0, 1.0f / (float)n_qkv);

  flash_attn_kernel<<<B_ * H_ * (N_ / 128), 256, 0, stream>>>(qb, kbf, vt, ao);

  proj_gemm_kernel<<<(1024 * 12) / 8, 256, 0, stream>>>(
      ao, tproj, b_proj, out, scales + 1, 1.0f / (float)n_proj);
}
